// LSTMDecoder_80041010528380
// MI455X (gfx1250) — compile-verified, matrix operand order checked by reading, not run on hardware
//
#include <hip/hip_runtime.h>

#define B_  256
#define D_  2048
#define L_  2
#define T_  32
#define ND  (4 * D_)   // 8192 gate width
#define KD  (2 * D_)   // 4096 fused K (x | h_prev)

typedef __attribute__((ext_vector_type(16))) __bf16 v16bf;
typedef __attribute__((ext_vector_type(8)))  __bf16 v8bf;
typedef __attribute__((ext_vector_type(8)))  float  v8f;

// ---- weight packing: fp32 -> bf16, fused [Wih | Whh], gate-interleaved rows.
// Output row n' = d*4 + gate  (gate 0..3 = i,f,g,o), so one GEMM block owns all
// four gates of a contiguous d-range and can apply the LSTM cell in-kernel.
__global__ __launch_bounds__(256) void pack_weights(const float* __restrict__ Wih,
                                                    const float* __restrict__ Whh,
                                                    __bf16* __restrict__ Wc) {
  size_t idx  = (size_t)blockIdx.x * blockDim.x + threadIdx.x;  // one quad each
  size_t base = idx * 4;
  size_t perL = (size_t)ND * KD;
  int    l    = (int)(base / perL);
  size_t r    = base - (size_t)l * perL;
  int    np   = (int)(r / KD);            // interleaved row
  int    k    = (int)(r - (size_t)np * KD);
  int    gate = np & 3;
  int    d    = np >> 2;
  int    n    = gate * D_ + d;            // original torch row
  const float* src = (k < D_) ? (Wih + ((size_t)l * ND + n) * D_ + k)
                              : (Whh + ((size_t)l * ND + n) * D_ + (k - D_));
  float4 v = *(const float4*)src;
  __bf16* dst = Wc + base;
  dst[0] = (__bf16)v.x; dst[1] = (__bf16)v.y;
  dst[2] = (__bf16)v.z; dst[3] = (__bf16)v.w;
}

// bias4[l][d*4+gate] = b_ih + b_hh (same interleaving as the weight rows)
__global__ __launch_bounds__(256) void pack_bias(const float* __restrict__ bih,
                                                 const float* __restrict__ bhh,
                                                 float* __restrict__ bias4) {
  int i = blockIdx.x * blockDim.x + threadIdx.x;
  if (i >= L_ * ND) return;
  int l = i / ND, r = i - l * ND;
  int d = r >> 2, gate = r & 3;
  int src = l * ND + gate * D_ + d;
  bias4[i] = bih[src] + bhh[src];
}

// ---------------- fused gates-GEMM + LSTM cell ------------------------------
#define BM 64     // block M tile
#define BN 128    // block N tile (= 32 d-values x 4 gates)
#define KK 64     // K step per LDS stage (2 WMMA K-fragments)

__global__ __launch_bounds__(256) void lstm_step(const float*  __restrict__ xsrc,  // [B,D]
                                                 const float*  __restrict__ hsrc,  // [B,D] h_prev
                                                 const __bf16* __restrict__ W,     // [ND,KD] interleaved
                                                 const float*  __restrict__ bias4, // [ND] interleaved
                                                 float*        __restrict__ cst,   // [B,D] in/out
                                                 float*        __restrict__ hst)   // [B,D] out
{
  __shared__ union {
    struct { __bf16 As[2][BM][KK]; __bf16 Ws[2][BN][KK]; } s;  // 48 KB staging
    float Cs[BM][BN];                                          // 32 KB epilogue
  } u;

  const int tid  = threadIdx.x;
  const int lane = tid & 31;
  const int wave = tid >> 5;     // 0..7
  const int wm   = wave >> 1;    // 0..3  -> M sub-tile (16 rows)
  const int wn   = wave & 1;     // 0..1  -> N sub-tile (64 cols)
  const int hi   = lane >> 4;    // lane half (ISA K-split / M-split)
  const int lm   = lane & 15;

  const int bm0 = blockIdx.y * BM;
  const int bn0 = blockIdx.x * BN;

  v8f acc[4] = {};

  // A tile: 64x64 fp32 -> bf16, 16 elements per thread (needs VALU cvt)
  auto loadA = [&](int buf, int kk) {
    const int row = tid >> 2;          // 0..63
    const int c0  = (tid & 3) * 16;    // 0..48
    const float* base = (kk < D_)
        ? (xsrc + (size_t)(bm0 + row) * D_ + kk + c0)
        : (hsrc + (size_t)(bm0 + row) * D_ + (kk - D_) + c0);
    float4 f0 = ((const float4*)base)[0];
    float4 f1 = ((const float4*)base)[1];
    float4 f2 = ((const float4*)base)[2];
    float4 f3 = ((const float4*)base)[3];
    alignas(16) __bf16 t[16];
    t[0]=(__bf16)f0.x;  t[1]=(__bf16)f0.y;  t[2]=(__bf16)f0.z;  t[3]=(__bf16)f0.w;
    t[4]=(__bf16)f1.x;  t[5]=(__bf16)f1.y;  t[6]=(__bf16)f1.z;  t[7]=(__bf16)f1.w;
    t[8]=(__bf16)f2.x;  t[9]=(__bf16)f2.y;  t[10]=(__bf16)f2.z; t[11]=(__bf16)f2.w;
    t[12]=(__bf16)f3.x; t[13]=(__bf16)f3.y; t[14]=(__bf16)f3.z; t[15]=(__bf16)f3.w;
    uint4* d = (uint4*)&u.s.As[buf][row][c0];
    d[0] = ((const uint4*)t)[0];
    d[1] = ((const uint4*)t)[1];
  };

  // W tile: 128x64 bf16, async DMA global -> LDS (ASYNCcnt, no VGPR round-trip)
  auto loadW = [&](int buf, int kk) {
    const int row = tid >> 1;          // 0..127
    const int cb  = (tid & 1) * 32;    // bf16 col: 0 or 32 (64B half-row)
    const __bf16* g = W + (size_t)(bn0 + row) * KD + kk + cb;
    unsigned lds = (unsigned)(size_t)&u.s.Ws[buf][row][cb];  // low32 == LDS offset
    asm volatile(
        "global_load_async_to_lds_b128 %0, %1, off\n\t"
        "global_load_async_to_lds_b128 %0, %1, off offset:16\n\t"
        "global_load_async_to_lds_b128 %0, %1, off offset:32\n\t"
        "global_load_async_to_lds_b128 %0, %1, off offset:48"
        :: "v"(lds), "v"(g) : "memory");
  };

  loadW(0, 0);
  loadA(0, 0);
  asm volatile("s_wait_asynccnt 0x0" ::: "memory");
  __syncthreads();

  int buf = 0;
  for (int kk = 0; kk < KD; kk += KK) {
    const int nk = kk + KK;
    if (nk < KD) { loadW(buf ^ 1, nk); loadA(buf ^ 1, nk); }

    // Load ALL fragments of this stage into distinct registers first so the
    // ds_loads pipeline instead of lock-stepping with the WMMAs.
    v16bf af[2];
    v16bf bf[2][4];
    #pragma unroll
    for (int kf = 0; kf < 2; kf++) {
      const __bf16* ar = &u.s.As[buf][wm * 16 + lm][kf * 32];
      v8bf a0 = *(const v8bf*)(ar + hi * 8);
      v8bf a1 = *(const v8bf*)(ar + 16 + hi * 8);
      af[kf] = __builtin_shufflevector(a0, a1, 0,1,2,3,4,5,6,7,8,9,10,11,12,13,14,15);
      #pragma unroll
      for (int nt = 0; nt < 4; nt++) {
        const __bf16* br = &u.s.Ws[buf][wn * 64 + nt * 16 + lm][kf * 32];
        v8bf b0 = *(const v8bf*)(br + hi * 8);
        v8bf b1 = *(const v8bf*)(br + 16 + hi * 8);
        bf[kf][nt] = __builtin_shufflevector(b0, b1, 0,1,2,3,4,5,6,7,8,9,10,11,12,13,14,15);
      }
    }
    #pragma unroll
    for (int kf = 0; kf < 2; kf++)
      #pragma unroll
      for (int nt = 0; nt < 4; nt++)
        acc[nt] = __builtin_amdgcn_wmma_f32_16x16x32_bf16(
            false, af[kf], false, bf[kf][nt], (short)0, acc[nt], false, false);

    asm volatile("s_wait_asynccnt 0x0" ::: "memory");
    __syncthreads();
    buf ^= 1;
  }

  // ---- epilogue: spill acc tile to LDS, re-bind threads (m,d)-major, apply cell
  // C/D layout: VGPR v -> M = v + 8*hi ; N = lane&15
  #pragma unroll
  for (int nt = 0; nt < 4; nt++) {
    const int n = wn * 64 + nt * 16 + lm;
    #pragma unroll
    for (int v = 0; v < 8; v++)
      u.Cs[wm * 16 + hi * 8 + v][n] = acc[nt][v];
  }
  __syncthreads();

  const float* bias_blk = bias4 + bn0;   // 128 interleaved bias values
  const int d0 = bn0 >> 2;               // first d owned by this block
  #pragma unroll
  for (int i = 0; i < 8; i++) {
    const int cix = tid + 256 * i;       // 0..2047 cells (64 m x 32 d)
    const int ml  = cix >> 5;
    const int dl  = cix & 31;
    float4 gv = *(const float4*)&u.Cs[ml][dl * 4];     // (i,f,g,o)
    float4 bv = *(const float4*)&bias_blk[dl * 4];
    const size_t off = (size_t)(bm0 + ml) * D_ + d0 + dl;
    float ig = gv.x + bv.x, fg = gv.y + bv.y, gg = gv.z + bv.z, og = gv.w + bv.w;
    float si = 1.f / (1.f + __expf(-ig));
    float sf = 1.f / (1.f + __expf(-fg));
    float so = 1.f / (1.f + __expf(-og));
    float c  = sf * cst[off] + si * tanhf(gg);
    float h  = so * tanhf(c);
    cst[off] = c;
    hst[off] = h;
  }
}

// ---------------- driver ----------------------------------------------------
extern "C" void kernel_launch(void* const* d_in, const int* in_sizes, int n_in,
                              void* d_out, int out_size, void* d_ws, size_t ws_size,
                              hipStream_t stream) {
  const float* h_in = (const float*)d_in[0];   // [B,D]
  const float* h0   = (const float*)d_in[1];   // [L,B,D]
  const float* c0   = (const float*)d_in[2];   // [L,B,D]
  const float* Wih  = (const float*)d_in[3];   // [L,ND,D]
  const float* Whh  = (const float*)d_in[4];   // [L,ND,D]
  const float* bih  = (const float*)d_in[5];   // [L,ND]
  const float* bhh  = (const float*)d_in[6];   // [L,ND]
  float* out = (float*)d_out;                  // [T,B,D]

  const size_t BD = (size_t)B_ * D_;
  char* ws = (char*)d_ws;
  __bf16* Wc   = (__bf16*)ws; ws += (size_t)L_ * ND * KD * sizeof(__bf16); // 128 MB (fits L2)
  float* bias4 = (float*)ws;  ws += (size_t)L_ * ND * sizeof(float);
  float* cst   = (float*)ws;  ws += (size_t)L_ * BD * sizeof(float);
  float* h0a   = (float*)ws;  ws += BD * sizeof(float);   // layer-0 h ping
  float* h0b   = (float*)ws;  ws += BD * sizeof(float);   // layer-0 h pong

  {
    size_t quads = (size_t)L_ * ND * KD / 4;
    pack_weights<<<(unsigned)(quads / 256), 256, 0, stream>>>(Wih, Whh, Wc);
    pack_bias<<<(L_ * ND + 255) / 256, 256, 0, stream>>>(bih, bhh, bias4);
  }
  hipMemcpyAsync(cst, c0, (size_t)L_ * BD * sizeof(float),
                 hipMemcpyDeviceToDevice, stream);

  float* h0buf[2] = { h0a, h0b };
  dim3 ggrid(ND / BN, B_ / BM);   // (64, 4) -> 256 blocks
  for (int t = 0; t < T_; t++) {
    // layer 0: x = previous final output (or initial h); h_prev ping-pongs
    const float* x0  = t ? (out + (size_t)(t - 1) * BD) : h_in;
    const float* hp0 = t ? h0buf[(t - 1) & 1] : h0;            // layer-0 init slice
    lstm_step<<<ggrid, 256, 0, stream>>>(x0, hp0, Wc, bias4, cst, h0buf[t & 1]);

    // layer 1: x = layer-0 h (just written); h_prev/h_new live in the output rows
    const float* hp1 = t ? (out + (size_t)(t - 1) * BD) : (h0 + BD);
    lstm_step<<<ggrid, 256, 0, stream>>>(h0buf[t & 1], hp1, Wc + (size_t)ND * KD,
                                         bias4 + ND, cst + BD, out + (size_t)t * BD);
  }
}
